// CLIPEncoderLayer_89541478187164
// MI455X (gfx1250) — matrix operand order repaired and checked by reading, not compiled, not run on hardware
//
#include <hip/hip_runtime.h>
#include <hip/hip_bf16.h>
#include <stdint.h>

// Problem dimensions (fixed by the reference)
#define B_  32
#define S_  257
#define E_  1024
#define H_  16
#define D_  64
#define I_  4096
#define M_  (B_ * S_)   // 8224 tokens

typedef __attribute__((ext_vector_type(16))) __bf16        bf16x16;
typedef __attribute__((ext_vector_type(8)))  float         v8f;
typedef __attribute__((ext_vector_type(4)))  unsigned int  v4u;
typedef __attribute__((ext_vector_type(8)))  unsigned int  v8u;
typedef unsigned short u16;

// fp32 -> bf16 (round-to-nearest-even), stored as raw u16
static __device__ __forceinline__ u16 f2bf(float f) {
    unsigned u = __float_as_uint(f);
    return (u16)((u + 0x7FFFu + ((u >> 16) & 1u)) >> 16);
}

// Assemble a 16-element bf16 WMMA operand from two 16-byte pieces
static __device__ __forceinline__ bf16x16 make_frag(v4u lo, v4u hi) {
    v8u w = __builtin_shufflevector(lo, hi, 0, 1, 2, 3, 4, 5, 6, 7);
    return __builtin_bit_cast(bf16x16, w);
}

static __device__ __forceinline__ float gelu_tanh(float x) {
    const float c = 0.7978845608028654f; // sqrt(2/pi)
    return 0.5f * x * (1.0f + tanhf(c * (x + 0.044715f * x * x * x)));
}

// CDNA5 async global->LDS copy (ASYNCcnt-tracked, no VGPR round-trip).
// Low 32 bits of a generic pointer to a __shared__ object are the LDS address.
static __device__ __forceinline__ void async_copy_b128(u16* lds_dst, const u16* gsrc) {
    unsigned lds_off = (unsigned)(size_t)lds_dst;
    unsigned long long ga = (unsigned long long)(size_t)gsrc;
    asm volatile("global_load_async_to_lds_b128 %0, %1, off"
                 :: "v"(lds_off), "v"(ga) : "memory");
}
static __device__ __forceinline__ void wait_asynccnt0() {
    asm volatile("s_wait_asynccnt 0x0" ::: "memory");
}

// ---------------------------------------------------------------------------
// fp32 -> bf16 conversion (weights)
// ---------------------------------------------------------------------------
__global__ void f32_to_bf16_kernel(const float* __restrict__ in,
                                   u16* __restrict__ out, int n) {
    int i = blockIdx.x * blockDim.x + threadIdx.x;
    int stride = gridDim.x * blockDim.x;
    for (; i < n; i += stride) out[i] = f2bf(in[i]);
}

// ---------------------------------------------------------------------------
// LayerNorm over 1024 cols, one block (256 threads) per row; bf16 output
// ---------------------------------------------------------------------------
__global__ __launch_bounds__(256) void layernorm_bf16_kernel(
        const float* __restrict__ x, const float* __restrict__ w,
        const float* __restrict__ b, u16* __restrict__ out) {
    int row = blockIdx.x;
    int t = threadIdx.x;
    const float4* xr = (const float4*)(x + (size_t)row * E_);
    float4 v = xr[t];
    float s = v.x + v.y + v.z + v.w;
    float q = v.x * v.x + v.y * v.y + v.z * v.z + v.w * v.w;
    #pragma unroll
    for (int d = 1; d < 32; d <<= 1) {
        s += __shfl_xor(s, d, 32);
        q += __shfl_xor(q, d, 32);
    }
    __shared__ float rs[8], rq[8];
    int wave = t >> 5, lane = t & 31;
    if (lane == 0) { rs[wave] = s; rq[wave] = q; }
    __syncthreads();
    s = 0.0f; q = 0.0f;
    #pragma unroll
    for (int i = 0; i < 8; ++i) { s += rs[i]; q += rq[i]; }
    float mean = s * (1.0f / E_);
    float var  = q * (1.0f / E_) - mean * mean;
    float rstd = rsqrtf(var + 1e-5f);
    u16* orow = out + (size_t)row * E_;
    int c0 = t * 4;
    float xv[4] = { v.x, v.y, v.z, v.w };
    #pragma unroll
    for (int e = 0; e < 4; ++e) {
        int c = c0 + e;
        orow[c] = f2bf((xv[e] - mean) * rstd * w[c] + b[c]);
    }
}

// ---------------------------------------------------------------------------
// bf16 WMMA GEMM: out = epilogue( A[M,K] @ W[N,K]^T + bias )
// 256 thr = 8 waves (4x2), block tile 128x128, wave tile 32x64, K-step 32.
// Double-buffered LDS, staged with global_load_async_to_lds_b128 (ASYNCcnt).
// OOB rows are clamped (not zeroed): their outputs are bounds-guarded, so
// EXEC stays all-ones (also required for WMMA).
// ---------------------------------------------------------------------------
__global__ __launch_bounds__(256) void gemm_bf16_kernel(
        const u16* __restrict__ A,          // [M,K] bf16
        const u16* __restrict__ W,          // [N,K] bf16 (torch Linear weight)
        const float* __restrict__ bias,     // [N]
        const float* __restrict__ residual, // [M,N] fp32 or nullptr
        float* __restrict__ outF,           // [M,N] fp32 or nullptr
        u16*  __restrict__ outB,            // [M,N] bf16 or nullptr
        int M, int N, int K, float out_scale, int act_gelu) {
    __shared__ u16 As[2][128 * 32];   // [buf][row][k]
    __shared__ u16 Ws[2][128 * 32];   // [buf][n][k]

    int tid = threadIdx.x;
    int wave = tid >> 5, lane = tid & 31;
    int g = lane >> 4, ln = lane & 15;
    int wm = wave & 3, wn = wave >> 2;       // 4 (M) x 2 (N) wave grid
    int m0 = blockIdx.y * 128, n0 = blockIdx.x * 128;

    // staging coordinates: 2 x 16B chunks of A and of W per thread
    int srow[2], skc[2];
    #pragma unroll
    for (int c = 0; c < 2; ++c) {
        int idx = tid + c * 256;
        srow[c] = idx >> 2;            // 0..127
        skc[c]  = (idx & 3) << 3;      // 0,8,16,24
    }

    auto stage = [&](int buf, int k0) {
        #pragma unroll
        for (int c = 0; c < 2; ++c) {
            int gm = m0 + srow[c]; if (gm > M - 1) gm = M - 1;
            int gn = n0 + srow[c]; if (gn > N - 1) gn = N - 1;
            async_copy_b128(&As[buf][srow[c] * 32 + skc[c]],
                            A + (size_t)gm * K + k0 + skc[c]);
            async_copy_b128(&Ws[buf][srow[c] * 32 + skc[c]],
                            W + (size_t)gn * K + k0 + skc[c]);
        }
    };

    v8f acc[2][4];
    #pragma unroll
    for (int mi = 0; mi < 2; ++mi)
        #pragma unroll
        for (int nj = 0; nj < 4; ++nj)
            acc[mi][nj] = {};

    int nk = K >> 5;
    stage(0, 0);
    for (int i = 0; i < nk; ++i) {
        wait_asynccnt0();      // my async fills of buf[i&1] have landed
        __syncthreads();       // all waves' fills landed; all done reading buf[(i+1)&1]
        if (i + 1 < nk) stage((i + 1) & 1, (i + 1) << 5);

        const u16* as = &As[i & 1][0];
        const u16* ws = &Ws[i & 1][0];
        bf16x16 af[2], bfr[4];
        #pragma unroll
        for (int mi = 0; mi < 2; ++mi) {  // A operand: lane=row; elems 0-7:K=8g.., 8-15:K=16+8g..
            const u16* p = as + (wm * 32 + mi * 16 + ln) * 32 + 8 * g;
            af[mi] = make_frag(*(const v4u*)p, *(const v4u*)(p + 16));
        }
        #pragma unroll
        for (int nj = 0; nj < 4; ++nj) {  // B operand: lane=col; elems 0-7:K=8g.., 8-15:K=16+8g..
            const u16* p = ws + (wn * 64 + nj * 16 + ln) * 32 + 8 * g;
            bfr[nj] = make_frag(*(const v4u*)p, *(const v4u*)(p + 16));
        }
        #pragma unroll
        for (int mi = 0; mi < 2; ++mi)
            #pragma unroll
            for (int nj = 0; nj < 4; ++nj)
                acc[mi][nj] = __builtin_amdgcn_wmma_f32_16x16x32_bf16(
                    false, af[mi], false, bfr[nj], (short)0, acc[mi][nj], false, false);
    }

    // epilogue: D layout -> lane holds one col; VGPR r holds row ..+r+8*(lane/16)
    #pragma unroll
    for (int mi = 0; mi < 2; ++mi) {
        #pragma unroll
        for (int nj = 0; nj < 4; ++nj) {
            int gn = n0 + wn * 64 + nj * 16 + ln;
            float bia = (gn < N) ? bias[gn] : 0.0f;
            #pragma unroll
            for (int r = 0; r < 8; ++r) {
                int gm = m0 + wm * 32 + mi * 16 + r + 8 * g;
                if (gm >= M || gn >= N) continue;
                float val = (acc[mi][nj][r] + bia) * out_scale;
                if (act_gelu) val = gelu_tanh(val);
                size_t oi = (size_t)gm * N + gn;
                if (residual) val += residual[oi];
                if (outF) outF[oi] = val;
                else      outB[oi] = f2bf(val);
            }
        }
    }
}

// ---------------------------------------------------------------------------
// Flash attention: grid (B*H, ceil(S/64)), 128 thr = 4 waves,
// each wave owns a 16-row Q tile; key tiles of 32, online softmax, bf16 WMMA.
// K tile staged via async-to-LDS; V tile transposed through VGPRs.
// ---------------------------------------------------------------------------
__global__ __launch_bounds__(128) void attention_kernel(
        const u16* __restrict__ q, const u16* __restrict__ k,
        const u16* __restrict__ v, const float* __restrict__ amask,
        const float* __restrict__ cmask, u16* __restrict__ ctx) {
    __shared__ u16 Kt[32 * 64];      // [key][d]
    __shared__ u16 Vt[64 * 32];      // [d][key] (transposed for B operand)
    __shared__ u16 Pt[4][16 * 32];   // per-wave P tile

    int bh = blockIdx.x;
    int b = bh >> 4, h = bh & 15;
    int tid = threadIdx.x;
    int wave = tid >> 5, lane = tid & 31;
    int g = lane >> 4, ln = lane & 15;
    int qbase = blockIdx.y * 64 + wave * 16;

    // Q fragments held in registers for the whole key loop (K-dim = d, 2 chunks)
    bf16x16 aq[2];
    {
        int sq = qbase + ln; if (sq > S_ - 1) sq = S_ - 1;
        const u16* qp = q + ((size_t)(b * S_ + sq)) * E_ + h * 64;
        #pragma unroll
        for (int c = 0; c < 2; ++c) {
            const u16* p = qp + c * 32 + 8 * g;
            aq[c] = make_frag(*(const v4u*)p, *(const v4u*)(p + 16));
        }
    }

    float mrun[8], lrun[8];
    v8f o[4];
    #pragma unroll
    for (int r = 0; r < 8; ++r) { mrun[r] = -1e30f; lrun[r] = 0.0f; }
    #pragma unroll
    for (int t = 0; t < 4; ++t) o[t] = {};

    const int NKT = (S_ + 31) / 32;  // 9 key tiles
    for (int kt = 0; kt < NKT; ++kt) {
        __syncthreads();             // everyone done reading Kt/Vt from prev iter
        #pragma unroll
        for (int it = 0; it < 2; ++it) {  // stage K tile (async) + V^T tile (VGPR transpose)
            int idx = tid + it * 128;     // 0..255: key = idx/8, dchunk = (idx%8)*8
            int key = idx >> 3, dc = (idx & 7) << 3;
            int kp = kt * 32 + key; if (kp > S_ - 1) kp = S_ - 1;
            size_t rowoff = ((size_t)(b * S_ + kp)) * E_ + h * 64 + dc;
            async_copy_b128(&Kt[key * 64 + dc], k + rowoff);
            v4u vv = *(const v4u*)(v + rowoff);
            #pragma unroll
            for (int e = 0; e < 4; ++e) {
                unsigned wv = vv[e];
                Vt[(dc + 2 * e) * 32 + key]     = (u16)(wv & 0xFFFFu);
                Vt[(dc + 2 * e + 1) * 32 + key] = (u16)(wv >> 16);
            }
        }
        wait_asynccnt0();
        __syncthreads();

        // scores: 16 q-rows x 32 keys, accumulate over d = 2 K-chunks
        v8f sc[2]; sc[0] = {}; sc[1] = {};
        #pragma unroll
        for (int c = 0; c < 2; ++c)
            #pragma unroll
            for (int j = 0; j < 2; ++j) {
                const u16* p = &Kt[(j * 16 + ln) * 64 + c * 32 + 8 * g];
                bf16x16 bk = make_frag(*(const v4u*)p, *(const v4u*)(p + 16));
                sc[j] = __builtin_amdgcn_wmma_f32_16x16x32_bf16(
                    false, aq[c], false, bk, (short)0, sc[j], false, false);
            }

        // add masks / bound invalid keys
        #pragma unroll
        for (int j = 0; j < 2; ++j) {
            int kp = kt * 32 + j * 16 + ln;
            bool kvalid = kp < S_;
            #pragma unroll
            for (int r = 0; r < 8; ++r) {
                int sq = qbase + r + 8 * g;
                if (kvalid && sq < S_) {
                    size_t mi = ((size_t)b * S_ + sq) * S_ + kp;
                    sc[j][r] += cmask[mi] + amask[mi];
                } else {
                    sc[j][r] = -1e30f;
                }
            }
        }

        // online softmax (row lives in a 16-lane half; xor<16 reduces per half)
        #pragma unroll
        for (int r = 0; r < 8; ++r) {
            float mx = fmaxf(sc[0][r], sc[1][r]);
            #pragma unroll
            for (int d = 1; d < 16; d <<= 1) mx = fmaxf(mx, __shfl_xor(mx, d, 32));
            float mnew = fmaxf(mrun[r], mx);
            float alpha = __expf(mrun[r] - mnew);
            float p0 = __expf(sc[0][r] - mnew);
            float p1 = __expf(sc[1][r] - mnew);
            sc[0][r] = p0; sc[1][r] = p1;
            float ps = p0 + p1;
            #pragma unroll
            for (int d = 1; d < 16; d <<= 1) ps += __shfl_xor(ps, d, 32);
            lrun[r] = lrun[r] * alpha + ps;
            mrun[r] = mnew;
            #pragma unroll
            for (int t = 0; t < 4; ++t) o[t][r] *= alpha;
        }

        // P (C-layout) -> LDS -> re-read in A-layout (wave-local, DS is in-order)
        u16* pw = &Pt[wave][0];
        #pragma unroll
        for (int j = 0; j < 2; ++j)
            #pragma unroll
            for (int r = 0; r < 8; ++r)
                pw[(r + 8 * g) * 32 + j * 16 + ln] = f2bf(sc[j][r]);
        bf16x16 ap;
        {
            const u16* p = &pw[ln * 32 + 8 * g];
            ap = make_frag(*(const v4u*)p, *(const v4u*)(p + 16));
        }
        #pragma unroll
        for (int t = 0; t < 4; ++t) {   // P @ V over 4 d-tiles of 16
            const u16* p = &Vt[(t * 16 + ln) * 32 + 8 * g];
            bf16x16 bv = make_frag(*(const v4u*)p, *(const v4u*)(p + 16));
            o[t] = __builtin_amdgcn_wmma_f32_16x16x32_bf16(
                false, ap, false, bv, (short)0, o[t], false, false);
        }
    }

    // write ctx (bf16, [token][h*64+d])
    #pragma unroll
    for (int r = 0; r < 8; ++r) {
        int sq = qbase + r + 8 * g;
        if (sq >= S_) continue;
        float inv = 1.0f / lrun[r];
        u16* crow = ctx + ((size_t)(b * S_ + sq)) * E_ + h * 64;
        #pragma unroll
        for (int t = 0; t < 4; ++t)
            crow[t * 16 + ln] = f2bf(o[t][r] * inv);
    }
}

// ---------------------------------------------------------------------------
extern "C" void kernel_launch(void* const* d_in, const int* in_sizes, int n_in,
                              void* d_out, int out_size, void* d_ws, size_t ws_size,
                              hipStream_t stream) {
    const float* hidden = (const float*)d_in[0];
    const float* amask  = (const float*)d_in[1];
    const float* cmask  = (const float*)d_in[2];
    const float* ln1w = (const float*)d_in[3];
    const float* ln1b = (const float*)d_in[4];
    const float* qw = (const float*)d_in[5];
    const float* qb = (const float*)d_in[6];
    const float* kw = (const float*)d_in[7];
    const float* kb = (const float*)d_in[8];
    const float* vw = (const float*)d_in[9];
    const float* vb = (const float*)d_in[10];
    const float* ow = (const float*)d_in[11];
    const float* ob = (const float*)d_in[12];
    const float* ln2w = (const float*)d_in[13];
    const float* ln2b = (const float*)d_in[14];
    const float* f1w = (const float*)d_in[15];
    const float* f1b = (const float*)d_in[16];
    const float* f2w = (const float*)d_in[17];
    const float* f2b = (const float*)d_in[18];
    float* out = (float*)d_out;

    char* ws = (char*)d_ws;
    size_t off = 0;
    auto carve = [&](size_t bytes) -> char* {
        char* p = ws + off;
        off += (bytes + 255) & ~(size_t)255;
        return p;
    };

    u16* wq_b  = (u16*)carve((size_t)E_ * E_ * 2);
    u16* wk_b  = (u16*)carve((size_t)E_ * E_ * 2);
    u16* wv_b  = (u16*)carve((size_t)E_ * E_ * 2);
    u16* wo_b  = (u16*)carve((size_t)E_ * E_ * 2);
    u16* wf1_b = (u16*)carve((size_t)I_ * E_ * 2);
    u16* wf2_b = (u16*)carve((size_t)E_ * I_ * 2);
    u16* xln   = (u16*)carve((size_t)M_ * E_ * 2);
    u16* qbuf  = (u16*)carve((size_t)M_ * E_ * 2);
    u16* kbuf  = (u16*)carve((size_t)M_ * E_ * 2);
    u16* vbuf  = (u16*)carve((size_t)M_ * E_ * 2);
    u16* ctxb  = (u16*)carve((size_t)M_ * E_ * 2);
    float* hbuf = (float*)carve((size_t)M_ * E_ * 4);
    u16* xln2  = (u16*)carve((size_t)M_ * E_ * 2);
    u16* actb  = (u16*)carve((size_t)M_ * (size_t)I_ * 2);

    // 1) weight conversion fp32 -> bf16
    f32_to_bf16_kernel<<<1024, 256, 0, stream>>>(qw,  wq_b,  E_ * E_);
    f32_to_bf16_kernel<<<1024, 256, 0, stream>>>(kw,  wk_b,  E_ * E_);
    f32_to_bf16_kernel<<<1024, 256, 0, stream>>>(vw,  wv_b,  E_ * E_);
    f32_to_bf16_kernel<<<1024, 256, 0, stream>>>(ow,  wo_b,  E_ * E_);
    f32_to_bf16_kernel<<<2048, 256, 0, stream>>>(f1w, wf1_b, I_ * E_);
    f32_to_bf16_kernel<<<2048, 256, 0, stream>>>(f2w, wf2_b, E_ * I_);

    // 2) LN1
    layernorm_bf16_kernel<<<M_, 256, 0, stream>>>(hidden, ln1w, ln1b, xln);

    // 3) QKV projections (q pre-scaled by D^-0.5 = 0.125 after bias, as in ref)
    dim3 gE(E_ / 128, (M_ + 127) / 128);
    gemm_bf16_kernel<<<gE, 256, 0, stream>>>(xln, wq_b, qb, nullptr, nullptr, qbuf,
                                             M_, E_, E_, 0.125f, 0);
    gemm_bf16_kernel<<<gE, 256, 0, stream>>>(xln, wk_b, kb, nullptr, nullptr, kbuf,
                                             M_, E_, E_, 1.0f, 0);
    gemm_bf16_kernel<<<gE, 256, 0, stream>>>(xln, wv_b, vb, nullptr, nullptr, vbuf,
                                             M_, E_, E_, 1.0f, 0);

    // 4) attention
    attention_kernel<<<dim3(B_ * H_, (S_ + 63) / 64), 128, 0, stream>>>(
        qbuf, kbuf, vbuf, amask, cmask, ctxb);

    // 5) O projection + residual -> h (fp32)
    gemm_bf16_kernel<<<gE, 256, 0, stream>>>(ctxb, wo_b, ob, hidden, hbuf, nullptr,
                                             M_, E_, E_, 1.0f, 0);

    // 6) LN2
    layernorm_bf16_kernel<<<M_, 256, 0, stream>>>(hbuf, ln2w, ln2b, xln2);

    // 7) fc1 + GELU(tanh)
    dim3 gI(I_ / 128, (M_ + 127) / 128);
    gemm_bf16_kernel<<<gI, 256, 0, stream>>>(xln2, wf1_b, f1b, nullptr, nullptr, actb,
                                             M_, I_, E_, 1.0f, 1);

    // 8) fc2 + residual -> output (fp32)
    gemm_bf16_kernel<<<gE, 256, 0, stream>>>(actb, wf2_b, f2b, hbuf, out, nullptr,
                                             M_, E_, I_, 1.0f, 0);
}
